// TradingTransformer_72945724555385
// MI455X (gfx1250) — matrix operand order repaired and checked by reading, not compiled, not run on hardware
//
#include <hip/hip_runtime.h>
#include <hip/hip_bf16.h>

// ---------------------------------------------------------------------------
// Transformer encoder-decoder forward for MI455X (gfx1250).
// All GEMMs (QKV/out projections, Q*K^T, attn*V, FFN) run on
// v_wmma_f32_16x16x32_bf16 (fp32 accumulate).  Compute-bound (AI ~2000
// flop/byte at 23.3 TB/s HBM); weights (112MB) are L2-resident (192MB L2).
//
// GEMM kernel: 256 threads = 8 wave32, block tile 128x128, BK=32.
// Waves tiled 4(M) x 2(N); each wave owns 32x64 = 2x4 wmma accumulators.
// Double-buffered LDS (one s_barrier per K-step); staging issues all
// global_load_b128s first (8 in flight), then v_cvt_pk_bf16_f32 + b64 LDS
// stores; global_prefetch of the K+2 tile.  Template <TRANSB, CHECK>: hot
// shapes are tile-aligned (branch-free); only attn*V (N=64) takes clamps.
// ---------------------------------------------------------------------------

typedef __attribute__((ext_vector_type(16))) __bf16          v16bf;
typedef __attribute__((ext_vector_type(2)))  __bf16          v2bf;
typedef __attribute__((ext_vector_type(8)))  float           v8f;

static constexpr int BM = 128;
static constexpr int BN = 128;
static constexpr int BK = 32;
static constexpr int LDS_S = BK + 4;   // 36 ushorts = 72B row stride: 8B-aligned
                                       // b64 stores, 18-word bank stride (no conflicts)

// fp32 -> bf16 via native hardware conversion (v_cvt_pk_bf16_f32 when available)
__device__ __forceinline__ unsigned pk2(float a, float b) {
#if __has_builtin(__builtin_amdgcn_cvt_pk_bf16_f32)
  auto p = __builtin_amdgcn_cvt_pk_bf16_f32(a, b);
  unsigned r;
  __builtin_memcpy(&r, &p, 4);
  return r;
#else
  v2bf p = { (__bf16)a, (__bf16)b };
  unsigned r;
  __builtin_memcpy(&r, &p, 4);
  return r;
#endif
}
__device__ __forceinline__ unsigned short f2bf(float f) {
  __bf16 h = (__bf16)f;
  return __builtin_bit_cast(unsigned short, h);
}

// ---- stage one [BM x BK] A tile and one [BN x BK] (transposed) B tile ----
// Gather phase first (all global_load_b128 in flight), then convert+store.
template <int TRANSB, bool CHECK>
__device__ __forceinline__ void stage_tiles(
    const float* __restrict__ A, const float* __restrict__ Bm,
    unsigned short* __restrict__ la, unsigned short* __restrict__ lb,
    int bm, int bn, int kt, int M, int N, int K, int lda, int ldb, int tid)
{
  if constexpr (!CHECK) {
    constexpr int AIT = (BM * BK) / (256 * 4);   // 4 float4 per thread
    constexpr int BIT = (BN * BK) / (256 * 4);   // 4 float4 per thread
    float4 fa[AIT], fb[BIT];
#pragma unroll
    for (int i = 0; i < AIT; ++i) {
      int e = tid + i * 256;
      int r = e >> 3, c = (e & 7) * 4;
      fa[i] = *(const float4*)(A + (long long)(bm + r) * lda + (kt + c));
    }
#pragma unroll
    for (int i = 0; i < BIT; ++i) {
      int e = tid + i * 256;
      if constexpr (TRANSB) {          // B is [N,K]: K-contiguous rows
        int n = e >> 3, c = (e & 7) * 4;
        fb[i] = *(const float4*)(Bm + (long long)(bn + n) * ldb + (kt + c));
      } else {                         // B is [K,N]: vector along N
        int kk = e >> 5, n0 = (e & 31) * 4;
        fb[i] = *(const float4*)(Bm + (long long)(kt + kk) * ldb + (bn + n0));
      }
    }
#pragma unroll
    for (int i = 0; i < AIT; ++i) {
      int e = tid + i * 256;
      int r = e >> 3, c = (e & 7) * 4;
      uint2 p; p.x = pk2(fa[i].x, fa[i].y); p.y = pk2(fa[i].z, fa[i].w);
      *(uint2*)(la + r * LDS_S + c) = p;
    }
#pragma unroll
    for (int i = 0; i < BIT; ++i) {
      int e = tid + i * 256;
      if constexpr (TRANSB) {
        int n = e >> 3, c = (e & 7) * 4;
        uint2 p; p.x = pk2(fb[i].x, fb[i].y); p.y = pk2(fb[i].z, fb[i].w);
        *(uint2*)(lb + n * LDS_S + c) = p;
      } else {                         // transpose into lb[n][k]
        int kk = e >> 5, n0 = (e & 31) * 4;
        lb[(n0 + 0) * LDS_S + kk] = f2bf(fb[i].x);
        lb[(n0 + 1) * LDS_S + kk] = f2bf(fb[i].y);
        lb[(n0 + 2) * LDS_S + kk] = f2bf(fb[i].z);
        lb[(n0 + 3) * LDS_S + kk] = f2bf(fb[i].w);
      }
    }
  } else {
    // Branch-free clamped scalar path (only attn*V: N=64 uses this)
    constexpr int AIT = (BM * BK) / 256;         // 16 scalars per thread
    constexpr int BIT = (BN * BK) / 256;
    float va[AIT], vb[BIT];
#pragma unroll
    for (int i = 0; i < AIT; ++i) {
      int idx = tid + i * 256;
      int r = idx >> 5, c = idx & 31;
      int gm = bm + r, gk = kt + c;
      int cm = gm < M ? gm : M - 1, ck = gk < K ? gk : K - 1;
      float vv = A[(long long)cm * lda + ck];
      va[i] = (gm < M && gk < K) ? vv : 0.f;
    }
#pragma unroll
    for (int i = 0; i < BIT; ++i) {
      int idx = tid + i * 256;
      int n = idx >> 5, c = idx & 31;
      int gn = bn + n, gk = kt + c;
      int cn = gn < N ? gn : N - 1, ck = gk < K ? gk : K - 1;
      long long off = TRANSB ? (long long)cn * ldb + ck : (long long)ck * ldb + cn;
      float vv = Bm[off];
      vb[i] = (gn < N && gk < K) ? vv : 0.f;
    }
#pragma unroll
    for (int i = 0; i < AIT; ++i) {
      int idx = tid + i * 256;
      int r = idx >> 5, c = idx & 31;
      la[r * LDS_S + c] = f2bf(va[i]);
    }
#pragma unroll
    for (int i = 0; i < BIT; ++i) {
      int idx = tid + i * 256;
      int n = idx >> 5, c = idx & 31;
      lb[n * LDS_S + c] = f2bf(vb[i]);
    }
  }
}

// ---- build fragments (ISA 7.12.2 layouts) and run 8 WMMAs ----
__device__ __forceinline__ void wmma_step(
    const unsigned short* __restrict__ la, const unsigned short* __restrict__ lb,
    int wm, int wn, int lr, int hi, v8f acc[2][4])
{
  typedef __attribute__((ext_vector_type(16))) unsigned short v16us;
  v16bf afrag[2], bfrag[4];
#pragma unroll
  for (int im = 0; im < 2; ++im) {
    int mrow = wm + im * 16 + lr;
    int kb = hi * 8;                 // lanes 0-15: K base 0 ; 16-31: K base 8
    v16us t;
#pragma unroll
    for (int p = 0; p < 8; ++p) {    // pair p -> elems 2p,2p+1 (K-contiguous)
      int kk = (p < 4) ? (kb + 2 * p) : (16 + kb + 2 * (p - 4));
      unsigned w = *(const unsigned*)&la[mrow * LDS_S + kk];
      t[2 * p]     = (unsigned short)(w & 0xffffu);
      t[2 * p + 1] = (unsigned short)(w >> 16);
    }
    afrag[im] = __builtin_bit_cast(v16bf, t);
  }
#pragma unroll
  for (int in = 0; in < 4; ++in) {
    int ncol = wn + in * 16 + lr;
    int kb = hi * 8;                 // lanes 0-15: K base 0 ; 16-31: K base 8
    v16us t;
#pragma unroll
    for (int p = 0; p < 8; ++p) {
      int kk = (p < 4) ? (kb + 2 * p) : (16 + kb + 2 * (p - 4));
      unsigned w = *(const unsigned*)&lb[ncol * LDS_S + kk];
      t[2 * p]     = (unsigned short)(w & 0xffffu);
      t[2 * p + 1] = (unsigned short)(w >> 16);
    }
    bfrag[in] = __builtin_bit_cast(v16bf, t);
  }
#pragma unroll
  for (int im = 0; im < 2; ++im)
#pragma unroll
    for (int in = 0; in < 4; ++in)
      acc[im][in] = __builtin_amdgcn_wmma_f32_16x16x32_bf16(
          false, afrag[im], false, bfrag[in], (short)0, acc[im][in],
          false, false);
}

// ---------------------------------------------------------------------------
// Batched GEMM: C[z] = act( alpha * A[z] (x) B[z] + bias )
// z decomposition: zo=z/zInner, zi=z%zInner; X += zo*sXo + zi*sXi.
// ---------------------------------------------------------------------------
template <int TRANSB, bool CHECK>
__global__ __launch_bounds__(256)
void wmma_gemm_kernel(const float* __restrict__ A, const float* __restrict__ Bm,
                      const float* __restrict__ bias, float* __restrict__ C,
                      int M, int N, int K, int lda, int ldb, int ldc,
                      float alpha, int relu, int zInner,
                      long long sAo, long long sAi,
                      long long sBo, long long sBi,
                      long long sCo, long long sCi)
{
  __shared__ unsigned short la[2][BM * LDS_S];   // bf16 bits, [m][k], double buffered
  __shared__ unsigned short lb[2][BN * LDS_S];   // bf16 bits, [n][k] (B^T layout)

  const int z  = blockIdx.z;
  const int zo = z / zInner, zi = z - zo * zInner;
  A  += zo * sAo + zi * sAi;
  Bm += zo * sBo + zi * sBi;
  C  += zo * sCo + zi * sCi;

  const int tid  = threadIdx.x;
  const int lane = tid & 31;
  const int wave = tid >> 5;
  const int wm   = (wave >> 1) * 32;   // 4 waves along M
  const int wn   = (wave & 1) * 64;    // 2 waves along N
  const int bm   = blockIdx.y * BM;
  const int bn   = blockIdx.x * BN;
  const int lr   = lane & 15;
  const int hi   = lane >> 4;

  v8f acc[2][4] = {};

  stage_tiles<TRANSB, CHECK>(A, Bm, la[0], lb[0], bm, bn, 0, M, N, K, lda, ldb, tid);

  int cur = 0;
  for (int kt = 0; kt < K; kt += BK) {
    __syncthreads();                               // buf[cur] ready for all waves
    const int ktn = kt + BK;
    if (ktn < K)
      stage_tiles<TRANSB, CHECK>(A, Bm, la[cur ^ 1], lb[cur ^ 1],
                                 bm, bn, ktn, M, N, K, lda, ldb, tid);
    if (kt + 2 * BK < K) {                         // prefetch K+2 tile into L2/L0
      __builtin_prefetch(A + (long long)(bm + (tid & (BM - 1))) * lda + kt + 2 * BK, 0, 1);
      if constexpr (TRANSB)
        __builtin_prefetch(Bm + (long long)(bn + (tid & (BN - 1))) * ldb + kt + 2 * BK, 0, 1);
      else
        __builtin_prefetch(Bm + (long long)(kt + 2 * BK + (tid & (BK - 1))) * ldb + bn, 0, 1);
    }
    wmma_step(la[cur], lb[cur], wm, wn, lr, hi, acc);
    cur ^= 1;
  }

  // ---- epilogue: alpha, bias, relu, store (C/D layout: lane->N, vgpr r->M) ----
#pragma unroll
  for (int im = 0; im < 2; ++im)
#pragma unroll
    for (int in = 0; in < 4; ++in) {
      int ncol = bn + wn + in * 16 + lr;
      float bv = 0.f;
      if (bias) {
        int cn = ncol < N ? ncol : N - 1;
        bv = bias[cn];
      }
#pragma unroll
      for (int r = 0; r < 8; ++r) {
        int mrow = bm + wm + im * 16 + r + hi * 8;
        float vv = acc[im][in][r] * alpha + bv;
        if (relu) vv = fmaxf(vv, 0.f);
        if constexpr (CHECK) {
          if (mrow < M && ncol < N)
            C[(long long)mrow * ldc + ncol] = vv;
        } else {
          C[(long long)mrow * ldc + ncol] = vv;
        }
      }
    }
}

// ---------------------------------------------------------------------------
// out[row] = LayerNorm(x[row] (+ res[row])) * g + b        (D = 512 fixed)
// ---------------------------------------------------------------------------
__global__ __launch_bounds__(256)
void ln_kernel(const float* __restrict__ x, const float* __restrict__ res,
               const float* __restrict__ g, const float* __restrict__ b,
               float* __restrict__ out)
{
  __shared__ float red[256];
  const long long row = blockIdx.x;
  const float* xr = x + row * 512;
  const float* rr = res ? res + row * 512 : nullptr;
  const int t = threadIdx.x;

  float v[2];
  float s = 0.f;
#pragma unroll
  for (int i = 0; i < 2; ++i) {
    int c = t + i * 256;
    float u = xr[c] + (rr ? rr[c] : 0.f);
    v[i] = u; s += u;
  }
  red[t] = s; __syncthreads();
  for (int o = 128; o > 0; o >>= 1) { if (t < o) red[t] += red[t + o]; __syncthreads(); }
  float mu = red[0] * (1.f / 512.f);
  __syncthreads();

  float s2 = 0.f;
#pragma unroll
  for (int i = 0; i < 2; ++i) { float d = v[i] - mu; s2 += d * d; }
  red[t] = s2; __syncthreads();
  for (int o = 128; o > 0; o >>= 1) { if (t < o) red[t] += red[t + o]; __syncthreads(); }
  float rstd = rsqrtf(red[0] * (1.f / 512.f) + 1e-5f);
  __syncthreads();

#pragma unroll
  for (int i = 0; i < 2; ++i) {
    int c = t + i * 256;
    out[row * 512 + c] = (v[i] - mu) * rstd * g[c] + b[c];
  }
}

// ---------------------------------------------------------------------------
// In-place row softmax over N=512, optional causal mask (k > q -> -1e9).
// ---------------------------------------------------------------------------
__global__ __launch_bounds__(256)
void softmax_kernel(float* __restrict__ s, int Lq, int causal)
{
  __shared__ float red[256];
  const long long row = blockIdx.x;
  const int q = (int)(row % Lq);
  float* sr = s + row * 512;
  const int t = threadIdx.x;

  float v[2];
  float mx = -1e30f;
#pragma unroll
  for (int i = 0; i < 2; ++i) {
    int c = t + i * 256;
    float u = sr[c];
    if (causal && c > q) u = -1e9f;
    v[i] = u; mx = fmaxf(mx, u);
  }
  red[t] = mx; __syncthreads();
  for (int o = 128; o > 0; o >>= 1) { if (t < o) red[t] = fmaxf(red[t], red[t + o]); __syncthreads(); }
  mx = red[0]; __syncthreads();

  float sum = 0.f;
#pragma unroll
  for (int i = 0; i < 2; ++i) { v[i] = __expf(v[i] - mx); sum += v[i]; }
  red[t] = sum; __syncthreads();
  for (int o = 128; o > 0; o >>= 1) { if (t < o) red[t] += red[t + o]; __syncthreads(); }
  float inv = 1.f / red[0]; __syncthreads();

#pragma unroll
  for (int i = 0; i < 2; ++i) sr[t + i * 256] = v[i] * inv;
}

// ---------------------------------------------------------------------------
// out = in + sinusoidal positional encoding (L x Dd), elementwise.
// ---------------------------------------------------------------------------
__global__ void posenc_add_kernel(const float* __restrict__ in, float* __restrict__ out,
                                  int L, int Dd, long long total)
{
  long long i = (long long)blockIdx.x * blockDim.x + threadIdx.x;
  if (i >= total) return;
  int d   = (int)(i % Dd);
  int pos = (int)((i / Dd) % L);
  int i2  = d & ~1;
  float div = __expf(-(float)i2 * (9.210340371976184f / (float)Dd)); // ln(1e4)
  float ang = (float)pos * div;
  float pe  = (d & 1) ? __cosf(ang) : __sinf(ang);
  out[i] = in[i] + pe;
}

// ============================ host orchestration ===========================

static void gemm(hipStream_t st, const float* A, const float* Bm, const float* bias,
                 float* C, int M, int N, int K, int lda, int ldb, int ldc,
                 int transB, float alpha, int relu,
                 int batch = 1, int zInner = 1,
                 long long sAo = 0, long long sAi = 0,
                 long long sBo = 0, long long sBi = 0,
                 long long sCo = 0, long long sCi = 0)
{
  dim3 grid((N + BN - 1) / BN, (M + BM - 1) / BM, batch);
  const bool full = (M % BM == 0) && (N % BN == 0);   // K is always a BK multiple here
  if (transB) {
    if (full)
      wmma_gemm_kernel<1, false><<<grid, 256, 0, st>>>(A, Bm, bias, C, M, N, K, lda, ldb, ldc,
                                                       alpha, relu, zInner, sAo, sAi, sBo, sBi, sCo, sCi);
    else
      wmma_gemm_kernel<1, true><<<grid, 256, 0, st>>>(A, Bm, bias, C, M, N, K, lda, ldb, ldc,
                                                      alpha, relu, zInner, sAo, sAi, sBo, sBi, sCo, sCi);
  } else {
    if (full)
      wmma_gemm_kernel<0, false><<<grid, 256, 0, st>>>(A, Bm, bias, C, M, N, K, lda, ldb, ldc,
                                                       alpha, relu, zInner, sAo, sAi, sBo, sBi, sCo, sCi);
    else
      wmma_gemm_kernel<0, true><<<grid, 256, 0, st>>>(A, Bm, bias, C, M, N, K, lda, ldb, ldc,
                                                      alpha, relu, zInner, sAo, sAi, sBo, sBi, sCo, sCi);
  }
}

static void layer_norm(hipStream_t st, const float* x, const float* res,
                       const float* g, const float* b, float* out, int rows)
{
  ln_kernel<<<rows, 256, 0, st>>>(x, res, g, b, out);
}

static void mha(hipStream_t st, const float* xq, const float* xkv, int Lq, int Lk,
                const float* W, const float* bias, int causal,
                float* q, float* k, float* v, float* scores, float* tmp, float* out)
{
  const int Bc = 8, Dd = 512, Hh = 8, Dk = 64;
  const int Mq = Bc * Lq, Mk = Bc * Lk;
  const long long DD = (long long)Dd * Dd;

  gemm(st, xq,  W + 0 * DD, bias + 0 * Dd, q, Mq, Dd, Dd, Dd, Dd, Dd, 0, 1.f, 0);
  gemm(st, xkv, W + 1 * DD, bias + 1 * Dd, k, Mk, Dd, Dd, Dd, Dd, Dd, 0, 1.f, 0);
  gemm(st, xkv, W + 2 * DD, bias + 2 * Dd, v, Mk, Dd, Dd, Dd, Dd, Dd, 0, 1.f, 0);

  // scores[b,h] = (1/sqrt(DK)) * Q[b,:,h,:] @ K[b,:,h,:]^T     (batched NT)
  gemm(st, q, k, nullptr, scores, Lq, Lk, Dk, Dd, Dd, Lk, 1, 0.125f, 0,
       Bc * Hh, Hh,
       (long long)Lq * Dd, Dk,
       (long long)Lk * Dd, Dk,
       (long long)Hh * Lq * Lk, (long long)Lq * Lk);

  softmax_kernel<<<Bc * Hh * Lq, 256, 0, st>>>(scores, Lq, causal);

  // tmp[b,:,h,:] = attn[b,h] @ V[b,:,h,:]                      (batched NN, N=64 -> CHECK)
  gemm(st, scores, v, nullptr, tmp, Lq, Dk, Lk, Lk, Dd, Dd, 0, 1.f, 0,
       Bc * Hh, Hh,
       (long long)Hh * Lq * Lk, (long long)Lq * Lk,
       (long long)Lk * Dd, Dk,
       (long long)Lq * Dd, Dk);

  gemm(st, tmp, W + 3 * DD, bias + 3 * Dd, out, Mq, Dd, Dd, Dd, Dd, Dd, 0, 1.f, 0);
}

static void ffn(hipStream_t st, const float* x, int M,
                const float* W1, const float* b1,
                const float* W2, const float* b2, float* mid, float* out)
{
  gemm(st, x,   W1, b1, mid, M, 2048, 512,  512,  2048, 2048, 0, 1.f, 1);
  gemm(st, mid, W2, b2, out, M, 512,  2048, 2048, 512,  512,  0, 1.f, 0);
}

extern "C" void kernel_launch(void* const* d_in, const int* in_sizes, int n_in,
                              void* d_out, int out_size, void* d_ws, size_t ws_size,
                              hipStream_t stream)
{
  const float* src  = (const float*)d_in[0];
  const float* tgt  = (const float*)d_in[1];
  const float* encW = (const float*)d_in[2];
  const float* encb = (const float*)d_in[3];
  const float* eW1  = (const float*)d_in[4];
  const float* eb1  = (const float*)d_in[5];
  const float* eW2  = (const float*)d_in[6];
  const float* eb2  = (const float*)d_in[7];
  const float* elng = (const float*)d_in[8];
  const float* elnb = (const float*)d_in[9];
  const float* efg  = (const float*)d_in[10];
  const float* efb  = (const float*)d_in[11];
  const float* dsW  = (const float*)d_in[12];
  const float* dsb  = (const float*)d_in[13];
  const float* dcW  = (const float*)d_in[14];
  const float* dcb  = (const float*)d_in[15];
  const float* dW1  = (const float*)d_in[16];
  const float* db1  = (const float*)d_in[17];
  const float* dW2  = (const float*)d_in[18];
  const float* db2  = (const float*)d_in[19];
  const float* dlng = (const float*)d_in[20];
  const float* dlnb = (const float*)d_in[21];
  const float* dfg  = (const float*)d_in[22];
  const float* dfb  = (const float*)d_in[23];
  // d_in[24] = tgt_mask (bool) : causal mask handled analytically in softmax

  const long long NT = 8LL * 512 * 512;   // 2M floats per [B,L,D] activation
  float* ws = (float*)d_ws;
  float* x      = ws; ws += NT;
  float* y      = ws; ws += NT;
  float* mem    = ws; ws += NT;
  float* q      = ws; ws += NT;
  float* k      = ws; ws += NT;
  float* v      = ws; ws += NT;
  float* t      = ws; ws += NT;
  float* ao     = ws; ws += NT;
  float* scores = ws; ws += 8LL * 8 * 512 * 512;   // [B,H,S,S]
  float* mid    = ws; ws += 8LL * 512 * 2048;      // [B,S,DFF]

  const int M = 8 * 512;
  const long long DD = 512LL * 512, DF = 512LL * 2048;

  posenc_add_kernel<<<(unsigned)((NT + 255) / 256), 256, 0, stream>>>(src, x, 512, 512, NT);
  posenc_add_kernel<<<(unsigned)((NT + 255) / 256), 256, 0, stream>>>(tgt, y, 512, 512, NT);

  // ----------------- encoder -----------------
  for (int l = 0; l < 4; ++l) {
    mha(stream, x, x, 512, 512, encW + (long long)l * 4 * DD, encb + (long long)l * 4 * 512, 0,
        q, k, v, scores, t, ao);
    layer_norm(stream, x, ao, elng + l * 2 * 512, elnb + l * 2 * 512, x, M);
    ffn(stream, x, M, eW1 + l * DF, eb1 + l * 2048, eW2 + l * DF, eb2 + l * 512, mid, ao);
    layer_norm(stream, x, ao, elng + l * 2 * 512 + 512, elnb + l * 2 * 512 + 512, x, M);
  }
  layer_norm(stream, x, nullptr, efg, efb, mem, M);

  // ----------------- decoder -----------------
  for (int l = 0; l < 4; ++l) {
    mha(stream, y, y, 512, 512, dsW + (long long)l * 4 * DD, dsb + (long long)l * 4 * 512, 1,
        q, k, v, scores, t, ao);
    layer_norm(stream, y, ao, dlng + l * 3 * 512, dlnb + l * 3 * 512, y, M);
    mha(stream, y, mem, 512, 512, dcW + (long long)l * 4 * DD, dcb + (long long)l * 4 * 512, 0,
        q, k, v, scores, t, ao);
    layer_norm(stream, y, ao, dlng + l * 3 * 512 + 512, dlnb + l * 3 * 512 + 512, y, M);
    ffn(stream, y, M, dW1 + l * DF, db1 + l * 2048, dW2 + l * DF, db2 + l * 512, mid, ao);
    layer_norm(stream, y, ao, dlng + l * 3 * 512 + 1024, dlnb + l * 3 * 512 + 1024, y, M);
  }
  layer_norm(stream, y, nullptr, dfg, dfb, (float*)d_out, M);
}
